// GrATiF_patch_18322330484755
// MI455X (gfx1250) — matrix operand order repaired and checked by reading, not compiled, not run on hardware
//
#include <hip/hip_runtime.h>
#include <hip/hip_bf16.h>
#include <stdint.h>

typedef __attribute__((ext_vector_type(16))) _Float16 v16h;
typedef __attribute__((ext_vector_type(8)))  float    v8f;

#if __has_builtin(__builtin_amdgcn_global_load_async_to_lds_b128) && \
    __has_builtin(__builtin_amdgcn_s_wait_asynccnt)
#define HAVE_ASYNC_LDS 1
typedef __attribute__((vector_size(16))) int i32x4;
typedef __attribute__((address_space(1))) i32x4 as1_i32x4;   // global
typedef __attribute__((address_space(3))) i32x4 as3_i32x4;   // LDS
#else
#define HAVE_ASYNC_LDS 0
#endif

namespace {
constexpr int R     = 16384;
constexpr int L     = 32;
constexpr int DIN   = 11;          // 1 + TE_DIM
constexpr int T     = 64;          // TTCN
constexpr int NC    = T * DIN;     // 704 columns of filt
constexpr int NT    = NC / 16;     // 44 N-tiles of 16 columns
constexpr int WAVES = 8;           // rows per block, one wave each
constexpr int BLOCK = WAVES * 32;
constexpr float NEGV = -100000000.0f;

constexpr int NPACK = NT * 2 * 32 * 16;                // packed f16 elements: 45056

// ---- dynamic LDS layout (bytes) ----
constexpr int SZ_W2P  = NPACK * 2;                     // 90112
constexpr int OFF_W2P = 0;
constexpr int OFF_B2  = OFF_W2P + SZ_W2P;              // f32[704]
constexpr int OFF_W1T = OFF_B2  + NC * 4;              // f32[64][12] transposed w_fg1
constexpr int OFF_B1  = OFF_W1T + T * 12 * 4;          // f32[64]
constexpr int OFF_TB  = OFF_B1  + T * 4;               // f32[64]
constexpr int OFF_SP  = OFF_TB  + T * 4;               // f32[20] scalar params
constexpr int OFF_XI  = (OFF_SP + 80 + 63) & ~63;      // f32[WAVES][32][12]
constexpr int OFF_HB  = OFF_XI  + WAVES * L * 12 * 4;  // f16[WAVES][32][64]
constexpr int OFF_ACC = OFF_HB  + WAVES * L * T * 2;   // f32[WAVES][64]
constexpr int LDS_SIZE = OFF_ACC + WAVES * T * 4;      // ~143.8 KB -> 2 WGs/WGP
static_assert(LDS_SIZE <= 160 * 1024, "two workgroups must fit a 320KB WGP");
} // namespace

// One-time: convert w_fg2 (64x704 f32) into f16, pre-packed in WMMA
// B-fragment order into workspace. packed p = (((nt*2+kt)*32 + lane)*16 + e),
// B order chosen here (32x16, KxN): lanes 0-15 hold K=0..15 (N=lane), lanes 16-31 K=16..31.
__global__ __launch_bounds__(256)
void gratif_pack_w2(const float* __restrict__ w_fg2, _Float16* __restrict__ w2p) {
    const int p = blockIdx.x * 256 + threadIdx.x;
    if (p >= NPACK) return;
    const int e  = p & 15;
    const int ln = (p >> 4) & 31;
    const int kt = (p >> 9) & 1;
    const int nt = p >> 10;
    const int k  = kt * 32 + ((ln >> 4) << 4) + e;
    const int n  = nt * 16 + (ln & 15);
    w2p[p] = (_Float16)w_fg2[k * NC + n];
}

__global__ __launch_bounds__(BLOCK, 2)
void gratif_fused(const float* __restrict__ x_time, const float* __restrict__ x_vals,
                  const float* __restrict__ x_mask,
                  const float* __restrict__ w_scale, const float* __restrict__ b_scale,
                  const float* __restrict__ w_per,   const float* __restrict__ b_per,
                  const float* __restrict__ w_fg1,   const float* __restrict__ b_fg1,
                  const float* __restrict__ w_fg2,   const float* __restrict__ b_fg2,
                  const float* __restrict__ t_bias,  float* __restrict__ out,
                  const _Float16* __restrict__ w2ws, int mode)
{
    extern __shared__ char smem[];
    _Float16* w2p  = (_Float16*)(smem + OFF_W2P);
    float*    bias2= (float*)(smem + OFF_B2);
    float*    w1t  = (float*)(smem + OFF_W1T);
    float*    b1   = (float*)(smem + OFF_B1);
    float*    tb   = (float*)(smem + OFF_TB);
    float*    sp   = (float*)(smem + OFF_SP);
    float*    xint = (float*)(smem + OFF_XI);
    _Float16* hb   = (_Float16*)(smem + OFF_HB);
    float*    accv = (float*)(smem + OFF_ACC);

    const int tid  = threadIdx.x;
    const int wave = tid >> 5;
    const int lane = tid & 31;
    const int hi   = lane >> 4;     // which half-wave
    const int lo   = lane & 15;
    const int row  = blockIdx.x * WAVES + wave;

    __builtin_prefetch(x_time + (size_t)blockIdx.x * WAVES * L, 0, 0);
    __builtin_prefetch(x_vals + (size_t)blockIdx.x * WAVES * L, 0, 0);

    // ---- stage packed f16 W2 into LDS ----
    if (mode) {
        // packed image already in workspace: bulk copy (async DMA if available,
        // overlapped with the x_int / first-layer computation below)
#if HAVE_ASYNC_LDS
        for (int p = tid; p < SZ_W2P / 16; p += BLOCK) {
            __builtin_amdgcn_global_load_async_to_lds_b128(
                (as1_i32x4*)(uintptr_t)((const char*)w2ws + p * 16),
                (as3_i32x4*)(unsigned)(uintptr_t)(smem + OFF_W2P + p * 16),
                0, 0);
        }
#else
        const uint4* s4 = (const uint4*)w2ws;
        uint4*       d4 = (uint4*)(smem + OFF_W2P);
        for (int p = tid; p < SZ_W2P / 16; p += BLOCK) d4[p] = s4[p];
#endif
    } else {
        // fallback: scattered in-kernel conversion straight from w_fg2
        for (int p = tid; p < NPACK; p += BLOCK) {
            const int e  = p & 15;
            const int ln = (p >> 4) & 31;
            const int kt = (p >> 9) & 1;
            const int nt = p >> 10;
            const int k  = kt * 32 + ((ln >> 4) << 4) + e;
            const int n  = nt * 16 + (ln & 15);
            w2p[p] = (_Float16)w_fg2[k * NC + n];
        }
    }

    // small parameter tables (w_fg1 stored transposed+padded: w1t[n][d])
    for (int i = tid; i < NC; i += BLOCK) bias2[i] = b_fg2[i];
    for (int i = tid; i < T * 12; i += BLOCK) {
        const int n = i / 12, d = i % 12;
        w1t[i] = (d < DIN) ? w_fg1[d * T + n] : 0.0f;
    }
    if (tid < T) { b1[tid] = b_fg1[tid]; tb[tid] = t_bias[tid]; }
    if (tid == 0) { sp[0] = w_scale[0]; sp[1] = b_scale[0]; }
    if (tid < 9)  { sp[2 + tid] = w_per[tid]; sp[11 + tid] = b_per[tid]; }
    accv[wave * T + lane]      = 0.0f;
    accv[wave * T + lane + 32] = 0.0f;
    __syncthreads();

    // ---- per-lane: one timestep l=lane of row `row` (overlaps async DMA) ----
    const float ttv = x_time[row * L + lane];
    const float vv  = x_vals[row * L + lane];
    const float mm  = x_mask[row * L + lane];
    const unsigned mbits = __builtin_amdgcn_ballot_w32(mm > 0.5f); // bit l = mask

    float xv[DIN];
    xv[0] = vv;
    xv[1] = fmaf(ttv, sp[0], sp[1]);
    #pragma unroll
    for (int j = 0; j < 9; ++j) xv[2 + j] = __sinf(fmaf(ttv, sp[2 + j], sp[11 + j]));
    #pragma unroll
    for (int d = 0; d < DIN; ++d) xint[(wave * L + lane) * 12 + d] = xv[d];

    // h = relu(x_int @ w_fg1 + b_fg1) -> f16 in LDS (A-matrix staging)
    _Float16* hrow = hb + (wave * L + lane) * T;
    for (int n = 0; n < T; ++n) {
        const float* wr = w1t + n * 12;      // contiguous: vector ds loads
        float a = b1[n];
        #pragma unroll
        for (int d = 0; d < DIN; ++d) a = fmaf(xv[d], wr[d], a);
        hrow[n] = (_Float16)fmaxf(a, 0.0f);
    }

#if HAVE_ASYNC_LDS
    __builtin_amdgcn_s_wait_asynccnt(0);     // W2 DMA landed in LDS
#endif
    __syncthreads();

    // ---- build persistent A fragments: h is M=32 x K=64 -> 2 Mtiles x 2 Ktiles
    // A order chosen here (16x32 f16, matches B): lane<16 row M=lane, elems 0-15 -> K=kb..kb+15;
    // lanes>=16 get K offset +16.
    union V16 { v16h v; _Float16 h[16]; };
    union V8  { v8f  v; float    f[8];  };
    V16 afr[2][2];
    #pragma unroll
    for (int mt = 0; mt < 2; ++mt) {
        const _Float16* src = hb + (wave * L + mt * 16 + lo) * T;
        #pragma unroll
        for (int kt = 0; kt < 2; ++kt)
            #pragma unroll
            for (int e = 0; e < 16; ++e)
                afr[mt][kt].h[e] = src[kt * 32 + hi * 16 + e];
    }

    // ---- stream 44 N-tiles: WMMA GEMM -> bias -> mask -> softmax(L) -> contract
    float* accw = accv + wave * T;
    int c = lo;                 // global filt column (incremental)
    int d = lo % DIN;           // x_int feature index
    int t = lo / DIN;           // output channel
    for (int nt = 0; nt < NT; ++nt) {
        v8f c0 = {}; v8f c1 = {};
        #pragma unroll
        for (int kt = 0; kt < 2; ++kt) {
            V16 bfr;
            bfr.v = *(const v16h*)(w2p + (((nt << 1) | kt) * 32 + lane) * 16);
            c0 = __builtin_amdgcn_wmma_f32_16x16x32_f16(false, afr[0][kt].v, false, bfr.v,
                                                        (short)0, c0, false, false);
            c1 = __builtin_amdgcn_wmma_f32_16x16x32_f16(false, afr[1][kt].v, false, bfr.v,
                                                        (short)0, c1, false, false);
        }
        V8 uc[2]; uc[0].v = c0; uc[1].v = c1;

        const float bias = bias2[c];

        // C layout: VGPR i holds M = mt*16 + hi*8 + i for column N = lo.
        float fm[16];
        float mx = NEGV;
        #pragma unroll
        for (int mt = 0; mt < 2; ++mt)
            #pragma unroll
            for (int i = 0; i < 8; ++i) {
                const int l = mt * 16 + hi * 8 + i;
                const float f = ((mbits >> l) & 1u) ? (uc[mt].f[i] + bias) : NEGV;
                fm[mt * 8 + i] = f;
                mx = fmaxf(mx, f);
            }
        mx = fmaxf(mx, __shfl_xor(mx, 16, 32));          // combine half-waves
        float s = 0.0f;
        #pragma unroll
        for (int j = 0; j < 16; ++j) { fm[j] = __expf(fm[j] - mx); s += fm[j]; }
        s += __shfl_xor(s, 16, 32);
        const float inv = __builtin_amdgcn_rcpf(s);

        float part = 0.0f;
        #pragma unroll
        for (int mt = 0; mt < 2; ++mt)
            #pragma unroll
            for (int i = 0; i < 8; ++i) {
                const int l = mt * 16 + hi * 8 + i;
                part = fmaf(fm[mt * 8 + i], xint[(wave * L + l) * 12 + d], part);
            }
        part *= inv;
        part += __shfl_xor(part, 16, 32);                // full sum over l
        if (hi == 0) atomicAdd(&accw[t], part);          // ds_add_f32

        // advance column by 16 = DIN + 5 (strength-reduced div/mod)
        c += 16; d += 5; ++t;
        if (d >= DIN) { d -= DIN; ++t; }
    }
    __syncthreads();

    // ---- epilogue: relu(acc + t_bias) ----
    {
        const float o0 = accw[lane]      + tb[lane];
        const float o1 = accw[lane + 32] + tb[lane + 32];
        out[row * T + lane]      = fmaxf(o0, 0.0f);
        out[row * T + lane + 32] = fmaxf(o1, 0.0f);
    }
}

extern "C" void kernel_launch(void* const* d_in, const int* in_sizes, int n_in,
                              void* d_out, int out_size, void* d_ws, size_t ws_size,
                              hipStream_t stream) {
    const float* x_time  = (const float*)d_in[0];
    const float* x_vals  = (const float*)d_in[1];
    const float* x_mask  = (const float*)d_in[2];
    const float* w_scale = (const float*)d_in[3];
    const float* b_scale = (const float*)d_in[4];
    const float* w_per   = (const float*)d_in[5];
    const float* b_per   = (const float*)d_in[6];
    const float* w_fg1   = (const float*)d_in[7];
    const float* b_fg1   = (const float*)d_in[8];
    const float* w_fg2   = (const float*)d_in[9];
    const float* b_fg2   = (const float*)d_in[10];
    const float* t_bias  = (const float*)d_in[11];
    float* out = (float*)d_out;

    const int mode = (ws_size >= (size_t)SZ_W2P) ? 1 : 0;
    if (mode) {
        gratif_pack_w2<<<(NPACK + 255) / 256, 256, 0, stream>>>(w_fg2, (_Float16*)d_ws);
    }

    (void)hipFuncSetAttribute((const void*)gratif_fused,
                              hipFuncAttributeMaxDynamicSharedMemorySize, LDS_SIZE);
    dim3 grid(R / WAVES);
    gratif_fused<<<grid, BLOCK, LDS_SIZE, stream>>>(
        x_time, x_vals, x_mask, w_scale, b_scale, w_per, b_per,
        w_fg1, b_fg1, w_fg2, b_fg2, t_bias, out, (const _Float16*)d_ws, mode);
}
